// AttentionConvolution_65163243815035
// MI455X (gfx1250) — matrix operand order repaired and checked by reading, not compiled, not run on hardware
//
#include <hip/hip_runtime.h>
#include <hip/hip_bf16.h>

typedef __attribute__((ext_vector_type(16))) __bf16    v16bf;
typedef __attribute__((ext_vector_type(8)))  float     v8f;
typedef __attribute__((ext_vector_type(4)))  unsigned  v4u;

union FragBF { v16bf bf; v4u u[2]; };

__device__ __forceinline__ unsigned short f2bf(float f) {
    union { float f; unsigned u; } x; x.f = f;
    unsigned r = x.u + 0x7FFFu + ((x.u >> 16) & 1u);
    return (unsigned short)(r >> 16);
}

__device__ __forceinline__ unsigned lds_off(const void* p) {
    return (unsigned)(unsigned long long)p;   // low 32 bits of flat shared addr = LDS offset
}

// CDNA5 async global->LDS copy (ASYNCcnt path). GVS addressing: saddr base + 32-bit vgpr offset.
__device__ __forceinline__ void async_ld_b128(unsigned lds_addr, const void* sbase, unsigned voff) {
    asm volatile("global_load_async_to_lds_b128 %0, %1, %2"
                 :: "v"(lds_addr), "v"(voff), "s"(sbase)
                 : "memory");
}
template <int N>
__device__ __forceinline__ void wait_async() {
    asm volatile("s_wait_asynccnt %0" :: "i"(N) : "memory");
}

#define BN 1024
#define DD 1024
#define HH 16
#define DH 64

// ---------------------------------------------------------------------------
// Kernel 1: fp32 -> bf16 conversions; weights transposed, Wq pre-scaled 1/32.
// ---------------------------------------------------------------------------
__global__ void cvt_kernel(const float* __restrict__ hid,
                           const float* __restrict__ Wq,
                           const float* __restrict__ Wk,
                           const float* __restrict__ Wv,
                           unsigned short* __restrict__ Abf,
                           unsigned short* __restrict__ Wqt,
                           unsigned short* __restrict__ Wkt,
                           unsigned short* __restrict__ Wvt) {
    size_t i = (size_t)blockIdx.x * 256 + threadIdx.x;
    Abf[i] = f2bf(hid[i]);
    if (i < (size_t)DD * DD) {
        int n = (int)(i >> 10), k = (int)(i & 1023);
        size_t src = (size_t)k * DD + n;                  // transpose: Wt[n][k] = W[k][n]
        Wqt[i] = f2bf(Wq[src] * 0.03125f);                // fold 1/sqrt(D)
        Wkt[i] = f2bf(Wk[src]);
        Wvt[i] = f2bf(Wv[src]);
    }
}

// ---------------------------------------------------------------------------
// Kernel 2: C[8192,1024] = Abf @ W  (W given transposed).  bf16 WMMA, LDS
// tiles double-buffered with async b128 copies (prefetch overlaps WMMA).
// z=0->Q, z=1->K (row-major bf16), z=2->V transposed per batch: Vt[b][d][n].
// ---------------------------------------------------------------------------
__global__ void __launch_bounds__(128)
qkv_gemm_kernel(const unsigned short* __restrict__ Abf,
                const unsigned short* __restrict__ Wqt,
                const unsigned short* __restrict__ Wkt,
                const unsigned short* __restrict__ Wvt,
                unsigned short* __restrict__ Qb,
                unsigned short* __restrict__ Kb,
                unsigned short* __restrict__ Vt) {
    __shared__ unsigned short shA[2][64 * 32];   // [m][k] rows of 64B, double buffered
    __shared__ unsigned short shB[2][64 * 32];   // [n][k] rows of 64B, double buffered
    const int tid  = threadIdx.x;
    const int wave = tid >> 5, lane = tid & 31;
    const int mrow = lane & 15, h2 = lane >> 4;
    const int m0 = blockIdx.x * 64, n0 = blockIdx.y * 64;
    const int which = blockIdx.z;
    const unsigned short* Wt = (which == 0) ? Wqt : (which == 1) ? Wkt : Wvt;

    const unsigned lA = lds_off(shA), lB = lds_off(shB);
    v8f acc[4] = {};

    // stage one 64x32 A tile + 64x32 B tile into buffer `buf` (4 async instr/wave)
    auto stage = [&](int buf, int kk) {
        #pragma unroll
        for (int c0 = 0; c0 < 2; ++c0) {
            int c = tid + c0 * 128;                       // 256 chunks of 16B each
            int row = c >> 2, ch = c & 3;
            unsigned offA = (unsigned)(((m0 + row) * DD + kk) * 2 + ch * 16);
            unsigned offB = (unsigned)(((n0 + row) * DD + kk) * 2 + ch * 16);
            async_ld_b128(lA + buf * 4096 + row * 64 + ch * 16, Abf, offA);
            async_ld_b128(lB + buf * 4096 + row * 64 + ch * 16, Wt,  offB);
        }
    };

    stage(0, 0);
    for (int i = 0; i < DD / 32; ++i) {
        const int buf = i & 1;
        const bool pre = (i + 1 < DD / 32);
        if (pre) {
            stage(buf ^ 1, (i + 1) * 32);   // prefetch next tile (overlaps this tile's WMMAs)
            wait_async<4>();                // current tile resident (prefetch still in flight)
        } else {
            wait_async<0>();
        }
        __syncthreads();

        FragBF a;
        const char* pA = (const char*)shA[buf] + (wave * 16 + mrow) * 64;
        a.u[0] = *(const v4u*)(pA + h2 * 32);
        a.u[1] = *(const v4u*)(pA + h2 * 32 + 16);
        #pragma unroll
        for (int nt = 0; nt < 4; ++nt) {
            FragBF b;
            const char* pB = (const char*)shB[buf] + (nt * 16 + mrow) * 64 + h2 * 32;
            b.u[0] = *(const v4u*)pB;
            b.u[1] = *(const v4u*)(pB + 16);
            acc[nt] = __builtin_amdgcn_wmma_f32_16x16x32_bf16(
                false, a.bf, false, b.bf, (short)0, acc[nt], false, false);
        }
        __syncthreads();   // frees `buf` for the prefetch issued next iteration
    }

    const int rbase = wave * 16 + 8 * h2;
    #pragma unroll
    for (int nt = 0; nt < 4; ++nt) {
        #pragma unroll
        for (int r = 0; r < 8; ++r) {
            int mg = m0 + rbase + r;
            int c  = n0 + nt * 16 + mrow;
            unsigned short hv = f2bf(acc[nt][r]);
            if (which == 0)      Qb[(size_t)mg * DD + c] = hv;
            else if (which == 1) Kb[(size_t)mg * DD + c] = hv;
            else {
                int bb = mg >> 10, key = mg & 1023;
                Vt[((size_t)bb << 20) + (size_t)c * BN + key] = hv;
            }
        }
    }
}

// ---------------------------------------------------------------------------
// Kernel 3: flash attention per (q-tile=64, head, batch). 4 waves, each wave
// owns 16 q rows. 32-key steps, double-buffered K/V^T LDS tiles: S via 4 WMMA,
// bias+mask, online softmax, P through LDS (C->A layout), P@V via 4 WMMA.
// ---------------------------------------------------------------------------
__global__ void __launch_bounds__(128)
attn_kernel(const unsigned short* __restrict__ Qb,
            const unsigned short* __restrict__ Kb,
            const unsigned short* __restrict__ Vt,
            const int* __restrict__ adj,
            const int* __restrict__ relpos,
            const float* __restrict__ rel_table,
            float* __restrict__ Ows) {
    __shared__ unsigned short shK[2][32 * 64];   // [key][d]  rows of 128B
    __shared__ unsigned short shV[2][64 * 32];   // [d][key]  rows of 64B
    __shared__ unsigned short shP[4][16 * 32];   // per-wave P, rows of 64B

    const int tid  = threadIdx.x;
    const int wave = tid >> 5, lane = tid & 31;
    const int mrow = lane & 15, h2 = lane >> 4;
    const int qt = blockIdx.x, h = blockIdx.y, b = blockIdx.z;
    const int q0 = qt * 64 + wave * 16;

    // Q A-fragments (d 0..31 and 32..63), row = q0 + mrow
    FragBF qf[2];
    {
        const char* qbase = (const char*)Qb +
            (((size_t)(b * BN + q0 + mrow)) * DD + h * DH) * 2;
        qf[0].u[0] = *(const v4u*)(qbase +       h2 * 16);
        qf[0].u[1] = *(const v4u*)(qbase + 32  + h2 * 16);
        qf[1].u[0] = *(const v4u*)(qbase + 64  + h2 * 16);
        qf[1].u[1] = *(const v4u*)(qbase + 96  + h2 * 16);
    }

    float mrun[8], lrun[8];
    v8f o[4] = {};
    #pragma unroll
    for (int r = 0; r < 8; ++r) { mrun[r] = -1e30f; lrun[r] = 0.f; }

    const unsigned lK = lds_off(shK), lV = lds_off(shV);

    // stage 32-key K tile + V^T tile into buffer `buf` (4 async instr/wave)
    auto stage = [&](int buf, int kt) {
        #pragma unroll
        for (int c0 = 0; c0 < 2; ++c0) {
            int c = tid + c0 * 128;                       // K: 32 keys x 128B
            int row = c >> 3, ch = c & 7;
            unsigned off = (unsigned)(((b * BN + kt + row) * DD + h * DH) * 2 + ch * 16);
            async_ld_b128(lK + buf * 4096 + row * 128 + ch * 16, Kb, off);
        }
        #pragma unroll
        for (int c0 = 0; c0 < 2; ++c0) {
            int c = tid + c0 * 128;                       // V^T: 64 d-rows x 64B
            int dd = c >> 2, ch = c & 3;
            unsigned off = ((unsigned)b << 21) +
                           (unsigned)(((h * DH + dd) * BN + kt) * 2 + ch * 16);
            async_ld_b128(lV + buf * 4096 + dd * 64 + ch * 16, Vt, off);
        }
    };

    stage(0, 0);
    for (int i = 0; i < BN / 32; ++i) {
        const int kt  = i * 32;
        const int buf = i & 1;
        const bool pre = (i + 1 < BN / 32);
        if (pre) {
            stage(buf ^ 1, kt + 32);        // prefetch next slab; overlaps compute below
            wait_async<4>();
        } else {
            wait_async<0>();
        }
        __syncthreads();

        // S = Q K^T : two 16-key tiles
        v8f s[2];
        #pragma unroll
        for (int j = 0; j < 2; ++j) {
            FragBF k0, k1;
            const char* kb = (const char*)shK[buf] + (j * 16 + mrow) * 128 + h2 * 16;
            k0.u[0] = *(const v4u*)kb;        k0.u[1] = *(const v4u*)(kb + 32);
            k1.u[0] = *(const v4u*)(kb + 64); k1.u[1] = *(const v4u*)(kb + 96);
            v8f z = {};
            z = __builtin_amdgcn_wmma_f32_16x16x32_bf16(
                    false, qf[0].bf, false, k0.bf, (short)0, z, false, false);
            s[j] = __builtin_amdgcn_wmma_f32_16x16x32_bf16(
                    false, qf[1].bf, false, k1.bf, (short)0, z, false, false);
        }

        // relative-position bias + adjacency mask (C layout: row r+8*h2, col mrow)
        #pragma unroll
        for (int j = 0; j < 2; ++j) {
            int kcol = kt + j * 16 + mrow;
            #pragma unroll
            for (int r = 0; r < 8; ++r) {
                size_t idx = ((size_t)(b * BN) + q0 + 8 * h2 + r) * BN + kcol;
                float bias = rel_table[relpos[idx]];
                s[j][r] = adj[idx] ? (s[j][r] + bias) : -1e9f;
            }
        }

        // online softmax over this 32-key slab
        float p0[8], p1[8];
        #pragma unroll
        for (int r = 0; r < 8; ++r) {
            float t = fmaxf(s[0][r], s[1][r]);
            #pragma unroll
            for (int off = 8; off > 0; off >>= 1) t = fmaxf(t, __shfl_xor(t, off, 32));
            float mn = fmaxf(mrun[r], t);
            float sc = __expf(mrun[r] - mn);
            mrun[r] = mn;
            p0[r] = __expf(s[0][r] - mn);
            p1[r] = __expf(s[1][r] - mn);
            float rs = p0[r] + p1[r];
            #pragma unroll
            for (int off = 8; off > 0; off >>= 1) rs += __shfl_xor(rs, off, 32);
            lrun[r] = lrun[r] * sc + rs;
            #pragma unroll
            for (int t4 = 0; t4 < 4; ++t4) o[t4][r] *= sc;
        }

        // P: C layout -> LDS -> A layout
        {
            char* pb = (char*)shP[wave];
            #pragma unroll
            for (int r = 0; r < 8; ++r) {
                int rr = r + 8 * h2;
                *(unsigned short*)(pb + rr * 64 + mrow * 2)        = f2bf(p0[r]);
                *(unsigned short*)(pb + rr * 64 + (16 + mrow) * 2) = f2bf(p1[r]);
            }
        }
        __syncthreads();

        FragBF pf;
        {
            const char* pr = (const char*)shP[wave] + mrow * 64;
            pf.u[0] = *(const v4u*)(pr + h2 * 32);
            pf.u[1] = *(const v4u*)(pr + h2 * 32 + 16);
        }
        #pragma unroll
        for (int t4 = 0; t4 < 4; ++t4) {
            FragBF vf;
            const char* vb = (const char*)shV[buf] + (t4 * 16 + mrow) * 64 + h2 * 32;
            vf.u[0] = *(const v4u*)vb;
            vf.u[1] = *(const v4u*)(vb + 16);
            o[t4] = __builtin_amdgcn_wmma_f32_16x16x32_bf16(
                        false, pf.bf, false, vf.bf, (short)0, o[t4], false, false);
        }
        __syncthreads();   // frees `buf` for next iteration's prefetch
    }

    #pragma unroll
    for (int t4 = 0; t4 < 4; ++t4) {
        #pragma unroll
        for (int r = 0; r < 8; ++r) {
            size_t idx = ((size_t)(b * BN) + q0 + 8 * h2 + r) * DD + h * DH + t4 * 16 + mrow;
            Ows[idx] = o[t4][r] / lrun[r];
        }
    }
}

// ---------------------------------------------------------------------------
// Kernel 4: relu + residual + layernorm, one row (D=1024) per 256-thread block
// ---------------------------------------------------------------------------
__global__ void __launch_bounds__(256)
ln_kernel(const float* __restrict__ Ows, const float* __restrict__ hid,
          const float* __restrict__ gamma, const float* __restrict__ beta,
          float* __restrict__ out) {
    __shared__ float red0[256];
    __shared__ float red1[256];
    const int row = blockIdx.x, tid = threadIdx.x;
    const float* o  = Ows + (size_t)row * DD;
    const float* hv = hid + (size_t)row * DD;
    float x[4], s = 0.f, ss = 0.f;
    #pragma unroll
    for (int i = 0; i < 4; ++i) {
        float t = fmaxf(o[tid * 4 + i], 0.f) + hv[tid * 4 + i];
        x[i] = t; s += t; ss += t * t;
    }
    red0[tid] = s; red1[tid] = ss;
    __syncthreads();
    for (int st = 128; st > 0; st >>= 1) {
        if (tid < st) { red0[tid] += red0[tid + st]; red1[tid] += red1[tid + st]; }
        __syncthreads();
    }
    float mean = red0[0] * (1.f / DD);
    float var  = red1[0] * (1.f / DD) - mean * mean;
    float rstd = rsqrtf(var + 1e-5f);
    #pragma unroll
    for (int i = 0; i < 4; ++i) {
        int c = tid * 4 + i;
        out[(size_t)row * DD + c] = (x[i] - mean) * rstd * gamma[c] + beta[c];
    }
}

// ---------------------------------------------------------------------------
extern "C" void kernel_launch(void* const* d_in, const int* in_sizes, int n_in,
                              void* d_out, int out_size, void* d_ws, size_t ws_size,
                              hipStream_t stream) {
    const float* hid  = (const float*)d_in[0];
    const int*   adj  = (const int*)d_in[1];
    const int*   rpos = (const int*)d_in[2];
    const float* Wq   = (const float*)d_in[3];
    const float* Wk   = (const float*)d_in[4];
    const float* Wv   = (const float*)d_in[5];
    const float* rtab = (const float*)d_in[6];
    const float* gam  = (const float*)d_in[7];
    const float* bet  = (const float*)d_in[8];

    char* ws = (char*)d_ws;
    const size_t MB = 1u << 20;
    unsigned short* Abf = (unsigned short*)(ws + 0 * MB);    // 16 MB
    unsigned short* Wqt = (unsigned short*)(ws + 16 * MB);   //  2 MB
    unsigned short* Wkt = (unsigned short*)(ws + 18 * MB);   //  2 MB
    unsigned short* Wvt = (unsigned short*)(ws + 20 * MB);   //  2 MB
    unsigned short* Qb  = (unsigned short*)(ws + 22 * MB);   // 16 MB
    unsigned short* Kb  = (unsigned short*)(ws + 38 * MB);   // 16 MB
    unsigned short* Vt  = (unsigned short*)(ws + 54 * MB);   // 16 MB
    float*          Ow  = (float*)         (ws + 70 * MB);   // 32 MB

    cvt_kernel<<<(8 * BN * DD) / 256, 256, 0, stream>>>(hid, Wq, Wk, Wv,
                                                        Abf, Wqt, Wkt, Wvt);
    dim3 gg(8 * BN / 64, DD / 64, 3);
    qkv_gemm_kernel<<<gg, 128, 0, stream>>>(Abf, Wqt, Wkt, Wvt, Qb, Kb, Vt);
    dim3 ga(BN / 64, HH, 8);
    attn_kernel<<<ga, 128, 0, stream>>>(Qb, Kb, Vt, adj, rpos, rtab, Ow);
    ln_kernel<<<8 * BN, 256, 0, stream>>>(Ow, hid, gam, bet, (float*)d_out);
}
